// gotermsClassifier_22608707846958
// MI455X (gfx1250) — matrix operand order repaired and checked by reading, not compiled, not run on hardware
//
#include <hip/hip_runtime.h>
#include <hip/hip_bf16.h>
#include <math.h>

typedef __attribute__((ext_vector_type(16))) __bf16 v16bf;
typedef __attribute__((ext_vector_type(8)))  __bf16 v8bf;
typedef __attribute__((ext_vector_type(4)))  __bf16 v4bf;
typedef __attribute__((ext_vector_type(8)))  float  v8f;

#define N_SEQ 2048
#define T_LEN 128
#define E_DIM 128
#define H_DIM 256
#define G_DIM 1024   // 4*H
#define L_SZ  1000

#define ROWS_PER_WG 32
#define LSTM_THREADS 256   // 8 waves; each wave: 32 gate-columns x both 16-row tiles

__device__ __forceinline__ float sigm(float x) { return 1.0f / (1.0f + __expf(-x)); }

// ---- A operand (16x32 bf16, row-major source with stride `stride` elems) ----
// A k order used here (contiguous; B matches): lanes 0-15 M=0..15 (K 0-15), lanes 16-31 same M (K 16-31).
__device__ __forceinline__ v16bf load_a_lds(const __bf16* base, int stride, int kbase,
                                            int hi, int m) {
    const __bf16* p = base + m * stride + kbase + hi * 16;
    v8bf lo = *(const v8bf*)(p);        // K = kb0 .. kb0+7
    v8bf hv = *(const v8bf*)(p + 8);    // K = kb0+8 .. kb0+15
    v16bf a;
#pragma unroll
    for (int e = 0; e < 8; ++e) { a[e] = lo[e]; a[e + 8] = hv[e]; }
    return a;
}

// ---- B operand (32x16 bf16). Lane L: column n=L%16, khalf=L/16, 16 contiguous k. ----
__device__ __forceinline__ v16bf load_b_glb(const __bf16* p) {
    v8bf lo = *(const v8bf*)(p);
    v8bf hv = *(const v8bf*)(p + 8);
    v16bf b;
#pragma unroll
    for (int e = 0; e < 8; ++e) { b[e] = lo[e]; b[e + 8] = hv[e]; }
    return b;
}

// ---------------- prep: f32 -> bf16 weights, zero accumulators ----------------
__global__ void prep_kernel(const float* __restrict__ W_ih, const float* __restrict__ W_hh,
                            __bf16* __restrict__ Wih_bf, __bf16* __restrict__ Whh_bf,
                            float* __restrict__ c_sum, float* __restrict__ loss_acc) {
    int i = blockIdx.x * blockDim.x + threadIdx.x;
    if (i < G_DIM * E_DIM) Wih_bf[i] = (__bf16)W_ih[i];
    if (i < G_DIM * H_DIM) Whh_bf[i] = (__bf16)W_hh[i];
    if (i < H_DIM) c_sum[i] = 0.0f;
    if (i == H_DIM) *loss_acc = 0.0f;
}

// ---------------- fused embed + input-proj + LSTM recurrence ----------------
__global__ __launch_bounds__(LSTM_THREADS) void lstm_kernel(
    const int* __restrict__ tokens, const int* __restrict__ lengths,
    const float* __restrict__ emb,
    const float* __restrict__ b_ih, const float* __restrict__ b_hh,
    const __bf16* __restrict__ Wih, const __bf16* __restrict__ Whh,
    float* __restrict__ h_f, float* __restrict__ c_sum) {

    __shared__ __align__(16) __bf16 x_buf[ROWS_PER_WG][E_DIM];  // 8 KB
    __shared__ __align__(16) __bf16 h_buf[ROWS_PER_WG][H_DIM];  // 16 KB
    __shared__ int lens[ROWS_PER_WG];

    const int tid   = threadIdx.x;
    const int lane  = tid & 31;
    const int w8    = tid >> 5;       // wave 0..7: gate-column slice u in [w8*32, w8*32+32)
    const int hi    = (lane >> 4) & 1;
    const int ln    = lane & 15;
    const int khalf = lane >> 4;
    const int n0    = blockIdx.x * ROWS_PER_WG;

    for (int i = tid; i < ROWS_PER_WG * H_DIM; i += LSTM_THREADS)
        ((__bf16*)h_buf)[i] = (__bf16)0.0f;
    if (tid < ROWS_PER_WG) lens[tid] = lengths[n0 + tid];
    __syncthreads();

    int len_r[2][8];
#pragma unroll
    for (int mt = 0; mt < 2; ++mt)
#pragma unroll
        for (int r = 0; r < 8; ++r) len_r[mt][r] = lens[mt * 16 + hi * 8 + r];

    // bias per (gate q, sub-tile s): column u = w8*32 + s*16 + ln ; gate idx g = q*256 + u
    float bias[4][2];
#pragma unroll
    for (int q = 0; q < 4; ++q)
#pragma unroll
        for (int s = 0; s < 2; ++s) {
            int g = q * H_DIM + w8 * 32 + s * 16 + ln;
            bias[q][s] = b_ih[g] + b_hh[g];
        }

    v8f c[2][2];     // cell state, [mt][s]
    v8f hreg[2][2];  // hidden state copy (for branchless masked update)
#pragma unroll
    for (int mt = 0; mt < 2; ++mt)
#pragma unroll
        for (int s = 0; s < 2; ++s)
#pragma unroll
            for (int e = 0; e < 8; ++e) { c[mt][s][e] = 0.0f; hreg[mt][s][e] = 0.0f; }

    const int colofs = w8 * 32 + ln;   // per-lane B column

    for (int t = 0; t < T_LEN; ++t) {
        // ---- stage x_t = bf16(emb[tokens[:, t]]) into LDS (float4 chunks) ----
        for (int ch = tid; ch < ROWS_PER_WG * (E_DIM / 4); ch += LSTM_THREADS) {
            int r = ch >> 5, k4 = ch & 31;   // 32 float4-chunks per row
            int tok = tokens[(n0 + r) * T_LEN + t];
            const float4 v = *(const float4*)(emb + tok * E_DIM + k4 * 4);
            v4bf b4;
            b4[0] = (__bf16)v.x; b4[1] = (__bf16)v.y;
            b4[2] = (__bf16)v.z; b4[3] = (__bf16)v.w;
            *(v4bf*)(&x_buf[r][k4 * 4]) = b4;
        }
        __syncthreads();

        // Opaque per-iteration *offset* (not pointer): keeps global address-space
        // provenance (global_load_b128, LOADcnt only) while preventing the
        // compiler from hoisting loop-invariant weight loads out of the t-loop.
        long zofs = 0;
        asm volatile("" : "+s"(zofs));
        const __bf16* wb_ih = Wih + zofs + colofs * E_DIM + khalf * 16;
        const __bf16* wb_hh = Whh + zofs + colofs * H_DIM + khalf * 16;

        // ---- gates = bias + x_t @ W_ih^T + h @ W_hh^T ----
        // acc[q][s][mt]: per-wave 32x32 output tile (32 gate cols x 32 rows)
        v8f acc[4][2][2];
#pragma unroll
        for (int q = 0; q < 4; ++q)
#pragma unroll
            for (int s = 0; s < 2; ++s)
#pragma unroll
                for (int mt = 0; mt < 2; ++mt)
#pragma unroll
                    for (int e = 0; e < 8; ++e) acc[q][s][mt][e] = bias[q][s];

#pragma unroll
        for (int kc = 0; kc < E_DIM / 32; ++kc) {              // input projection, K=128
            v16bf a0 = load_a_lds(&x_buf[0][0],  E_DIM, kc * 32, hi, ln);
            v16bf a1 = load_a_lds(&x_buf[16][0], E_DIM, kc * 32, hi, ln);
            v16bf bb[2];                                       // depth-2 lookahead
            bb[0] = load_b_glb(wb_ih + kc * 32);
            bb[1] = load_b_glb(wb_ih + 16 * E_DIM + kc * 32);
#pragma unroll
            for (int i = 0; i < 8; ++i) {                      // i = q*2+s
                v16bf bc = bb[i & 1];
                if (i + 2 < 8) {
                    int qn = (i + 2) >> 1, sn = (i + 2) & 1;
                    bb[i & 1] = load_b_glb(wb_ih + (qn * H_DIM + sn * 16) * E_DIM + kc * 32);
                }
                int q = i >> 1, s = i & 1;
                acc[q][s][0] = __builtin_amdgcn_wmma_f32_16x16x32_bf16(
                    false, a0, false, bc, (short)0, acc[q][s][0], false, false);
                acc[q][s][1] = __builtin_amdgcn_wmma_f32_16x16x32_bf16(
                    false, a1, false, bc, (short)0, acc[q][s][1], false, false);
            }
        }
#pragma unroll
        for (int kc = 0; kc < H_DIM / 32; ++kc) {              // hidden projection, K=256
            v16bf a0 = load_a_lds(&h_buf[0][0],  H_DIM, kc * 32, hi, ln);
            v16bf a1 = load_a_lds(&h_buf[16][0], H_DIM, kc * 32, hi, ln);
            v16bf bb[2];
            bb[0] = load_b_glb(wb_hh + kc * 32);
            bb[1] = load_b_glb(wb_hh + 16 * H_DIM + kc * 32);
#pragma unroll
            for (int i = 0; i < 8; ++i) {
                v16bf bc = bb[i & 1];
                if (i + 2 < 8) {
                    int qn = (i + 2) >> 1, sn = (i + 2) & 1;
                    bb[i & 1] = load_b_glb(wb_hh + (qn * H_DIM + sn * 16) * H_DIM + kc * 32);
                }
                int q = i >> 1, s = i & 1;
                acc[q][s][0] = __builtin_amdgcn_wmma_f32_16x16x32_bf16(
                    false, a0, false, bc, (short)0, acc[q][s][0], false, false);
                acc[q][s][1] = __builtin_amdgcn_wmma_f32_16x16x32_bf16(
                    false, a1, false, bc, (short)0, acc[q][s][1], false, false);
            }
        }
        __syncthreads();   // all waves done reading h_buf/x_buf

        // ---- branchless cell update (mask via selects, no exec divergence) ----
#pragma unroll
        for (int mt = 0; mt < 2; ++mt)
#pragma unroll
            for (int s = 0; s < 2; ++s) {
                int u = w8 * 32 + s * 16 + ln;
#pragma unroll
                for (int r = 0; r < 8; ++r) {
                    bool m = (t < len_r[mt][r]);
                    float iv = sigm(acc[0][s][mt][r]);
                    float fv = sigm(acc[1][s][mt][r]);
                    float gv = tanhf(acc[2][s][mt][r]);
                    float ov = sigm(acc[3][s][mt][r]);
                    float cn = fv * c[mt][s][r] + iv * gv;
                    float hn = ov * tanhf(cn);
                    c[mt][s][r]    = m ? cn : c[mt][s][r];
                    hreg[mt][s][r] = m ? hn : hreg[mt][s][r];
                    h_buf[mt * 16 + hi * 8 + r][u] = (__bf16)hreg[mt][s][r];
                }
            }
        __syncthreads();
    }

    // ---- epilogue: accumulate mean(c) and emit h_f ----
#pragma unroll
    for (int s = 0; s < 2; ++s) {
        int u = w8 * 32 + s * 16 + ln;
        float sum = 0.0f;
#pragma unroll
        for (int mt = 0; mt < 2; ++mt)
#pragma unroll
            for (int r = 0; r < 8; ++r) sum += c[mt][s][r];
        atomicAdd(&c_sum[u], sum);
    }
    for (int i = tid; i < ROWS_PER_WG * H_DIM; i += LSTM_THREADS) {
        int r = i >> 8, u = i & (H_DIM - 1);
        h_f[(n0 + r) * H_DIM + u] = (float)h_buf[r][u];
    }
}

// ---------------- logits + log_softmax + NLL, one WG per sequence ----------------
__global__ __launch_bounds__(256) void loss_kernel(
    const float* __restrict__ h_f, const float* __restrict__ W_out,
    const float* __restrict__ b_out, const int* __restrict__ labels,
    float* __restrict__ loss_acc) {
    __shared__ float hrow[H_DIM];
    __shared__ float logits[L_SZ];
    __shared__ float red[256];
    const int n = blockIdx.x, tid = threadIdx.x;

    hrow[tid] = h_f[n * H_DIM + tid];   // blockDim == H_DIM == 256
    __syncthreads();

    for (int l = tid; l < L_SZ; l += 256) {
        float acc = b_out[l];
        const float* w = W_out + l * H_DIM;
#pragma unroll 8
        for (int k = 0; k < H_DIM; ++k) acc = fmaf(hrow[k], w[k], acc);
        logits[l] = acc;
    }
    __syncthreads();

    float mx = -INFINITY;
    for (int l = tid; l < L_SZ; l += 256) mx = fmaxf(mx, logits[l]);
    red[tid] = mx;
    __syncthreads();
    for (int s = 128; s > 0; s >>= 1) {
        if (tid < s) red[tid] = fmaxf(red[tid], red[tid + s]);
        __syncthreads();
    }
    mx = red[0];
    __syncthreads();

    float se = 0.0f;
    for (int l = tid; l < L_SZ; l += 256) se += __expf(logits[l] - mx);
    red[tid] = se;
    __syncthreads();
    for (int s = 128; s > 0; s >>= 1) {
        if (tid < s) red[tid] += red[tid + s];
        __syncthreads();
    }
    if (tid == 0) {
        float lse = mx + __logf(red[0]);
        float lp  = logits[labels[n]] - lse;
        atomicAdd(loss_acc, -lp / (float)N_SEQ);
    }
}

// ---------------- finalize: d_out = [mean(c) (256), loss (1)] ----------------
__global__ void finalize_kernel(const float* __restrict__ c_sum,
                                const float* __restrict__ loss_acc,
                                float* __restrict__ out) {
    int i = threadIdx.x;
    if (i < H_DIM) out[i] = c_sum[i] / (float)N_SEQ;
    if (i == H_DIM) out[H_DIM] = *loss_acc;
}

extern "C" void kernel_launch(void* const* d_in, const int* in_sizes, int n_in,
                              void* d_out, int out_size, void* d_ws, size_t ws_size,
                              hipStream_t stream) {
    (void)in_sizes; (void)n_in; (void)out_size; (void)ws_size;
    const int*   tokens  = (const int*)d_in[0];
    const int*   lengths = (const int*)d_in[1];
    const int*   labels  = (const int*)d_in[2];
    const float* emb     = (const float*)d_in[3];
    const float* W_ih    = (const float*)d_in[4];
    const float* W_hh    = (const float*)d_in[5];
    const float* b_ih    = (const float*)d_in[6];
    const float* b_hh    = (const float*)d_in[7];
    const float* W_out   = (const float*)d_in[8];
    const float* b_out   = (const float*)d_in[9];
    float* out = (float*)d_out;

    char* ws = (char*)d_ws;
    __bf16* Wih_bf   = (__bf16*)(ws);                       // 256 KB
    __bf16* Whh_bf   = (__bf16*)(ws + (256 << 10));         // 512 KB
    float*  c_sum    = (float*)(ws + (768 << 10));          // 1 KB
    float*  loss_acc = (float*)(ws + (768 << 10) + 1024);   // 4 B
    float*  h_f      = (float*)(ws + (1 << 20));            // 2 MB

    prep_kernel<<<(G_DIM * H_DIM + 255) / 256, 256, 0, stream>>>(
        W_ih, W_hh, Wih_bf, Whh_bf, c_sum, loss_acc);

    lstm_kernel<<<N_SEQ / ROWS_PER_WG, LSTM_THREADS, 0, stream>>>(
        tokens, lengths, emb, b_ih, b_hh, Wih_bf, Whh_bf, h_f, c_sum);

    loss_kernel<<<N_SEQ, 256, 0, stream>>>(h_f, W_out, b_out, labels, loss_acc);

    finalize_kernel<<<1, 320, 0, stream>>>(c_sum, loss_acc, out);
}
